// CosineSimCausalTransformer_34926674051311
// MI455X (gfx1250) — compile-verified, matrix operand order checked by reading, not run on hardware
//
#include <hip/hip_runtime.h>
#include <math.h>

// ---------------------------------------------------------------- types ----
typedef __bf16 bf16_t;
typedef __bf16 v16bf __attribute__((ext_vector_type(16)));
typedef __bf16 v8bf  __attribute__((ext_vector_type(8)));
typedef float  v8f   __attribute__((ext_vector_type(8)));

#define BB     2
#define NN     2048
#define DD     512
#define VV     32000
#define HH     8
#define DHH    64
#define DFFF   2048
#define DEPTHH 6
#define A2     (BB * NN)            // 4096 rows total
#define RES_SCALE_C 1.8612097182041991f   // (2*6)^0.25
#define EPS_LN_C 1e-5f
#define EPS_L2_C 1e-12f
#define SCALE_C 8.0f

// load one 16-bit A/B WMMA fragment (ISA layout: lanes 0-15 hold K=0..7 /
// 16..23, lanes 16-31 hold K=8..15 / 24..31, two bf16 per VGPR)
__device__ __forceinline__ v16bf ldfrag16(const bf16_t* p) {
  v8bf lo = *(const v8bf*)p;
  v8bf hi = *(const v8bf*)(p + 16);
  return __builtin_shufflevector(lo, hi, 0, 1, 2, 3, 4, 5, 6, 7,
                                 8, 9, 10, 11, 12, 13, 14, 15);
}

// ---------------------------------------------------- transpose + cvt -----
// out[c*rows + r] = (bf16) in[r*cols + c];  32x32 LDS tile, block=256
__global__ __launch_bounds__(256)
void k_transpose_cvt(const float* __restrict__ in, bf16_t* __restrict__ out,
                     int rows, int cols) {
  __shared__ float tile[32][33];
  const int bc = blockIdx.x * 32;     // col base
  const int br = blockIdx.y * 32;     // row base
  const int tx = threadIdx.x & 31;
  const int ty = threadIdx.x >> 5;    // 0..7
#pragma unroll
  for (int i = 0; i < 32; i += 8)
    tile[ty + i][tx] = in[(size_t)(br + ty + i) * cols + (bc + tx)];
  __syncthreads();
#pragma unroll
  for (int i = 0; i < 32; i += 8)
    out[(size_t)(bc + ty + i) * rows + (br + tx)] = (bf16_t)tile[tx][ty + i];
}

// ----------------------------------------------------------- embedding ----
__global__ __launch_bounds__(256)
void k_embed(const int* __restrict__ x, const float* __restrict__ tok,
             const float* __restrict__ pos, float* __restrict__ hf,
             bf16_t* __restrict__ hb) {
  const int row = blockIdx.x;               // 0..A2-1
  const int n   = row & (NN - 1);
  const int t   = x[row];
#pragma unroll
  for (int j = threadIdx.x; j < DD; j += 256) {
    float v = tok[(size_t)t * DD + j] + pos[(size_t)n * DD + j];
    hf[(size_t)row * DD + j] = v;
    hb[(size_t)row * DD + j] = (bf16_t)v;
  }
}

// ------------------------------------------------------- WMMA bf16 GEMM ---
// C[M x N] (f32) = A[M x K] (bf16, row-major) * Bt[N x K]^T (bf16, row-major)
// wave: 32x64 tile (2 A frags x 4 B frags = 8 WMMAs per K-step).
// K loop unrolled x2 with ping-pong fragment buffers (no pipeline copies).
// block: 8 waves as 4(M) x 2(N) -> 128x128 tile.
// Requires M%128==0, N%128==0, K%64==0.
__global__ __launch_bounds__(256)
void k_gemm_bf16(const bf16_t* __restrict__ A, const bf16_t* __restrict__ Bt,
                 float* __restrict__ C, int M, int N, int K) {
  const int lane = threadIdx.x & 31;
  const int wave = threadIdx.x >> 5;
  const int wm = wave & 3, wn = wave >> 2;
  const int m0 = blockIdx.y * 128 + wm * 32;
  const int n0 = blockIdx.x * 128 + wn * 64;
  const int r    = lane & 15;
  const int half = lane >> 4;

  const bf16_t* aR0 = A + (size_t)(m0 + r) * K + half * 8;
  const bf16_t* aR1 = A + (size_t)(m0 + 16 + r) * K + half * 8;
  const bf16_t* bR  = Bt + (size_t)(n0 + r) * K + half * 8;
  const size_t bs = (size_t)16 * K;

  v8f acc[2][4] = {};
  v16bf a0[2], a1[2], b0[4], b1[4];

  a0[0] = ldfrag16(aR0);
  a0[1] = ldfrag16(aR1);
#pragma unroll
  for (int t = 0; t < 4; ++t) b0[t] = ldfrag16(bR + t * bs);

  for (int kt = 0; kt < K; kt += 64) {
    // prefetch buffer 1 @ kt+32
    a1[0] = ldfrag16(aR0 + kt + 32);
    a1[1] = ldfrag16(aR1 + kt + 32);
#pragma unroll
    for (int t = 0; t < 4; ++t) b1[t] = ldfrag16(bR + t * bs + kt + 32);
    // compute buffer 0
#pragma unroll
    for (int mi = 0; mi < 2; ++mi)
#pragma unroll
      for (int t = 0; t < 4; ++t)
        acc[mi][t] = __builtin_amdgcn_wmma_f32_16x16x32_bf16(
            false, a0[mi], false, b0[t], (short)0, acc[mi][t], false, false);
    // prefetch buffer 0 @ kt+64 (dummy reload of 0 on the last iteration)
    const int ktn = (kt + 64 < K) ? kt + 64 : 0;
    a0[0] = ldfrag16(aR0 + ktn);
    a0[1] = ldfrag16(aR1 + ktn);
#pragma unroll
    for (int t = 0; t < 4; ++t) b0[t] = ldfrag16(bR + t * bs + ktn);
    // compute buffer 1
#pragma unroll
    for (int mi = 0; mi < 2; ++mi)
#pragma unroll
      for (int t = 0; t < 4; ++t)
        acc[mi][t] = __builtin_amdgcn_wmma_f32_16x16x32_bf16(
            false, a1[mi], false, b1[t], (short)0, acc[mi][t], false, false);
  }
  // C/D layout: VGPR v, lane L -> row = m0 + mi*16 + v + (L>=16)*8,
  //                               col = n0 + t*16 + L%16
#pragma unroll
  for (int mi = 0; mi < 2; ++mi)
#pragma unroll
    for (int t = 0; t < 4; ++t)
#pragma unroll
      for (int v = 0; v < 8; ++v)
        C[(size_t)(m0 + mi * 16 + half * 8 + v) * N + (n0 + t * 16 + r)] =
            acc[mi][t][v];
}

// --------------------------------------------- pack q/k/v for attention ---
// qkv: [A2][1536] f32 (q|k|v). Emits (per head):
//   qb [BH][N][64] bf16 = l2norm(q) * 8      (A fragments)
//   kb [BH][N][64] bf16 = l2norm(k)          (Bt fragments for q@k^T)
//   vt [BH][64][N] bf16 = v transposed       (Bt fragments for P@V)
__global__ __launch_bounds__(256)
void k_pack_qkv(const float* __restrict__ qkv, bf16_t* __restrict__ qb,
                bf16_t* __restrict__ kb, bf16_t* __restrict__ vt) {
  __shared__ float vls[32][65];
  const int bh = blockIdx.y;            // 0..15
  const int b  = bh >> 3, h = bh & 7;
  const int n0 = blockIdx.x * 32;
  const int nl  = threadIdx.x >> 3;     // 0..31 row within tile
  const int sub = threadIdx.x & 7;      // 8 threads per row, 8 elems each

  const size_t srow = (size_t)(b * NN + n0 + nl) * 1536 + h * DHH + sub * 8;
  const size_t drow = ((size_t)bh * NN + n0 + nl) * DHH + sub * 8;

  // --- q: l2 normalize, fold in SCALE ---
  float qv[8], ss = 0.f;
#pragma unroll
  for (int t = 0; t < 8; ++t) { qv[t] = qkv[srow + t]; ss += qv[t] * qv[t]; }
  ss += __shfl_xor(ss, 1, 32); ss += __shfl_xor(ss, 2, 32); ss += __shfl_xor(ss, 4, 32);
  float inv = SCALE_C / fmaxf(sqrtf(ss), EPS_L2_C);
#pragma unroll
  for (int t = 0; t < 8; ++t) qb[drow + t] = (bf16_t)(qv[t] * inv);

  // --- k: l2 normalize ---
  ss = 0.f;
#pragma unroll
  for (int t = 0; t < 8; ++t) { qv[t] = qkv[srow + 512 + t]; ss += qv[t] * qv[t]; }
  ss += __shfl_xor(ss, 1, 32); ss += __shfl_xor(ss, 2, 32); ss += __shfl_xor(ss, 4, 32);
  inv = 1.0f / fmaxf(sqrtf(ss), EPS_L2_C);
#pragma unroll
  for (int t = 0; t < 8; ++t) kb[drow + t] = (bf16_t)(qv[t] * inv);

  // --- v: transpose 32(n) x 64(d) tile through LDS ---
#pragma unroll
  for (int t = 0; t < 8; ++t)
    vls[nl][sub * 8 + t] = qkv[srow + 1024 + t];
  __syncthreads();
  const int d = threadIdx.x >> 2;       // 0..63
  const int g = threadIdx.x & 3;        // 4 threads per d-row, 8 n each
#pragma unroll
  for (int t = 0; t < 8; ++t)
    vt[((size_t)bh * DHH + d) * NN + n0 + g * 8 + t] = (bf16_t)vls[g * 8 + t][d];
}

// ------------------------------------------ WMMA flash causal attention ---
// one wave per 16-query tile; 32-key chunks; S and O via WMMA; P repacked
// C-layout -> A-layout through per-wave LDS scratch.
__global__ __launch_bounds__(256)
void k_flash_attn(const bf16_t* __restrict__ qb, const bf16_t* __restrict__ kb,
                  const bf16_t* __restrict__ vt, bf16_t* __restrict__ ob) {
  __shared__ float pls[8][16 * 32];
  const int lane = threadIdx.x & 31;
  const int wave = threadIdx.x >> 5;
  int task = blockIdx.x * 8 + wave;         // BH * (N/16) = 2048 tasks
  task = __builtin_amdgcn_readfirstlane(task);
  const int qt = task & 127;                // N/16 - 1
  const int bh = task >> 7;
  const int i0 = qt * 16;
  const int r    = lane & 15;
  const int half = lane >> 4;
  float* pw = pls[wave];

  const bf16_t* qrow = qb + ((size_t)bh * NN + i0 + r) * DHH + half * 8;
  const v16bf aq0 = ldfrag16(qrow);         // K = 0..31
  const v16bf aq1 = ldfrag16(qrow + 32);    // K = 32..63
  const bf16_t* kbase = kb + (size_t)bh * NN * DHH;
  const bf16_t* vbase = vt + (size_t)bh * DHH * NN;

  float mrow[8], lrow[8];
  v8f o[4] = {};
#pragma unroll
  for (int v = 0; v < 8; ++v) { mrow[v] = -3.0e38f; lrow[v] = 0.f; }

  for (int j0 = 0; j0 < i0 + 16; j0 += 32) {
    // ---- S = q @ k^T (16 x 32), 2 j-subtiles x 2 K-steps ----
    v8f s[2];
#pragma unroll
    for (int jt = 0; jt < 2; ++jt) {
      const bf16_t* kr = kbase + (size_t)(j0 + jt * 16 + r) * DHH + half * 8;
      v8f z = {};
      z = __builtin_amdgcn_wmma_f32_16x16x32_bf16(false, aq0, false, ldfrag16(kr),
                                                  (short)0, z, false, false);
      z = __builtin_amdgcn_wmma_f32_16x16x32_bf16(false, aq1, false, ldfrag16(kr + 32),
                                                  (short)0, z, false, false);
      s[jt] = z;
    }
    // ---- causal mask (only chunks touching the diagonal) ----
    if (j0 + 31 > i0) {
#pragma unroll
      for (int jt = 0; jt < 2; ++jt)
#pragma unroll
        for (int v = 0; v < 8; ++v) {
          const int ri = i0 + v + half * 8;
          const int ci = j0 + jt * 16 + r;
          s[jt][v] = (ci > ri) ? -3.0e38f : s[jt][v];
        }
    }
    // ---- online softmax (row reductions within 16-lane half) ----
#pragma unroll
    for (int v = 0; v < 8; ++v) {
      float rm = fmaxf(s[0][v], s[1][v]);
      rm = fmaxf(rm, __shfl_xor(rm, 1, 32));
      rm = fmaxf(rm, __shfl_xor(rm, 2, 32));
      rm = fmaxf(rm, __shfl_xor(rm, 4, 32));
      rm = fmaxf(rm, __shfl_xor(rm, 8, 32));
      const float mn   = fmaxf(mrow[v], rm);
      const float corr = __expf(mrow[v] - mn);
      mrow[v] = mn;
      const float p0 = __expf(s[0][v] - mn);
      const float p1 = __expf(s[1][v] - mn);
      float rs = p0 + p1;
      rs += __shfl_xor(rs, 1, 32);
      rs += __shfl_xor(rs, 2, 32);
      rs += __shfl_xor(rs, 4, 32);
      rs += __shfl_xor(rs, 8, 32);
      lrow[v] = lrow[v] * corr + rs;
      s[0][v] = p0; s[1][v] = p1;
#pragma unroll
      for (int dt = 0; dt < 4; ++dt) o[dt][v] *= corr;
    }
    // ---- P: C-layout -> LDS -> A-fragment (bf16) ----
#pragma unroll
    for (int v = 0; v < 8; ++v) {
      pw[(v + half * 8) * 32 + r]      = s[0][v];
      pw[(v + half * 8) * 32 + 16 + r] = s[1][v];
    }
    asm volatile("s_wait_dscnt 0x0" ::: "memory");
    v16bf ap;
    {
      const float* pr = pw + r * 32 + half * 8;
#pragma unroll
      for (int t = 0; t < 8; ++t) {
        ap[t]     = (bf16_t)pr[t];
        ap[t + 8] = (bf16_t)pr[16 + t];
      }
    }
    // ---- O += P @ V (16 x 64), vt rows are Bt fragments ----
#pragma unroll
    for (int dt = 0; dt < 4; ++dt) {
      const bf16_t* vr = vbase + (size_t)(dt * 16 + r) * NN + j0 + half * 8;
      o[dt] = __builtin_amdgcn_wmma_f32_16x16x32_bf16(false, ap, false, ldfrag16(vr),
                                                      (short)0, o[dt], false, false);
    }
    asm volatile("s_wait_dscnt 0x0" ::: "memory");   // WAR: next chunk rewrites pw
  }
  // ---- epilogue: /= l, store bf16 into [A2][512] ----
  const int b = bh >> 3, h = bh & 7;
#pragma unroll
  for (int v = 0; v < 8; ++v) {
    const float invl = 1.0f / lrow[v];
    const int row = b * NN + i0 + v + half * 8;
#pragma unroll
    for (int dt = 0; dt < 4; ++dt)
      ob[(size_t)row * DD + h * DHH + dt * 16 + r] = (bf16_t)(o[dt][v] * invl);
  }
}

// ----------------------------------------------------------- gelu + cvt ---
__global__ __launch_bounds__(256)
void k_gelu_cvt(const float* __restrict__ in, bf16_t* __restrict__ out) {
  const size_t idx = (size_t)blockIdx.x * 256 + threadIdx.x;
  float x = in[idx];
  float r = 0.5f * x * (1.0f + erff(x * 0.70710678118654752f));
  out[idx] = (bf16_t)r;
}

// -------------------------------------- residual + layernorm (fused) ------
__global__ __launch_bounds__(256)
void k_residual_ln(const float* __restrict__ add, float* __restrict__ hf,
                   bf16_t* __restrict__ hb, const float* __restrict__ g,
                   const float* __restrict__ bta) {
  __shared__ float red[8];
  const int row = blockIdx.x;
  const int tid = threadIdx.x;
  const size_t base = (size_t)row * DD;
  float v0 = add[base + tid]       + hf[base + tid] * RES_SCALE_C;
  float v1 = add[base + tid + 256] + hf[base + tid + 256] * RES_SCALE_C;

  float s = v0 + v1;
#pragma unroll
  for (int m = 16; m >= 1; m >>= 1) s += __shfl_xor(s, m, 32);
  if ((tid & 31) == 0) red[tid >> 5] = s;
  __syncthreads();
  float mean = (red[0] + red[1] + red[2] + red[3] +
                red[4] + red[5] + red[6] + red[7]) * (1.0f / DD);
  float d0 = v0 - mean, d1 = v1 - mean;
  float sv = d0 * d0 + d1 * d1;
#pragma unroll
  for (int m = 16; m >= 1; m >>= 1) sv += __shfl_xor(sv, m, 32);
  __syncthreads();
  if ((tid & 31) == 0) red[tid >> 5] = sv;
  __syncthreads();
  float var = (red[0] + red[1] + red[2] + red[3] +
               red[4] + red[5] + red[6] + red[7]) * (1.0f / DD);
  float inv = rsqrtf(var + EPS_LN_C);
  float r0 = d0 * inv * g[tid] + bta[tid];
  float r1 = d1 * inv * g[tid + 256] + bta[tid + 256];
  hf[base + tid]       = r0;
  hf[base + tid + 256] = r1;
  hb[base + tid]       = (bf16_t)r0;
  hb[base + tid + 256] = (bf16_t)r1;
}

// ------------------------------------------------------------- launcher ---
extern "C" void kernel_launch(void* const* d_in, const int* in_sizes, int n_in,
                              void* d_out, int out_size, void* d_ws, size_t ws_size,
                              hipStream_t stream) {
  const int*   x    = (const int*)d_in[0];
  const float* tok  = (const float*)d_in[1];
  const float* pos  = (const float*)d_in[2];
  const float* Wq   = (const float*)d_in[3];
  const float* Wk   = (const float*)d_in[4];
  const float* Wv   = (const float*)d_in[5];
  const float* Wo   = (const float*)d_in[6];
  const float* ln1g = (const float*)d_in[7];
  const float* ln1b = (const float*)d_in[8];
  const float* W1   = (const float*)d_in[9];
  const float* W2   = (const float*)d_in[10];
  const float* ln2g = (const float*)d_in[11];
  const float* ln2b = (const float*)d_in[12];
  const float* Wl   = (const float*)d_in[13];
  float* out = (float*)d_out;

  // ---- workspace arena ----
  char* ws = (char*)d_ws;
  size_t off = 0;
  auto alloc = [&](size_t bytes) { char* p = ws + off; off += bytes; return p; };
  bf16_t* qkvT = (bf16_t*)alloc((size_t)DEPTHH * 1536 * 512 * 2);
  bf16_t* woT  = (bf16_t*)alloc((size_t)DEPTHH * 512 * 512 * 2);
  bf16_t* w1T  = (bf16_t*)alloc((size_t)DEPTHH * 2048 * 512 * 2);
  bf16_t* w2T  = (bf16_t*)alloc((size_t)DEPTHH * 512 * 2048 * 2);
  bf16_t* wlT  = (bf16_t*)alloc((size_t)VV * 512 * 2);
  float*  hf   = (float*)alloc((size_t)A2 * DD * 4);
  bf16_t* hb   = (bf16_t*)alloc((size_t)A2 * DD * 2);
  float*  tmp  = (float*)alloc((size_t)A2 * DD * 4);
  char*   scr  = alloc((size_t)A2 * DFFF * 4 + (size_t)A2 * DFFF * 2); // union
  // phase A (attention)
  float*  qkv   = (float*)scr;                                   // A2*1536 f32
  bf16_t* qb    = (bf16_t*)(scr + (size_t)A2 * 1536 * 4);        // 16*2048*64 bf16
  bf16_t* kbf   = (bf16_t*)(scr + (size_t)A2 * 1536 * 4 + (size_t)1 * 4194304);
  bf16_t* vtb   = (bf16_t*)(scr + (size_t)A2 * 1536 * 4 + (size_t)2 * 4194304);
  bf16_t* attnb = (bf16_t*)(scr + (size_t)A2 * 1536 * 4 + (size_t)3 * 4194304);
  // phase B (FFN) — reuses same region (phase A dead by then)
  float*  ff1 = (float*)scr;                                     // A2*2048 f32
  bf16_t* ffb = (bf16_t*)(scr + (size_t)A2 * DFFF * 4);          // A2*2048 bf16

  // ---- weight transpose + bf16 conversion ----
  for (int i = 0; i < DEPTHH; ++i) {
    dim3 g16(16, 16);   // 512x512
    k_transpose_cvt<<<g16, 256, 0, stream>>>(Wq + (size_t)i * 512 * 512,
                                             qkvT + (size_t)i * 1536 * 512, 512, 512);
    k_transpose_cvt<<<g16, 256, 0, stream>>>(Wk + (size_t)i * 512 * 512,
                                             qkvT + (size_t)i * 1536 * 512 + 512 * 512, 512, 512);
    k_transpose_cvt<<<g16, 256, 0, stream>>>(Wv + (size_t)i * 512 * 512,
                                             qkvT + (size_t)i * 1536 * 512 + 1024 * 512, 512, 512);
    k_transpose_cvt<<<g16, 256, 0, stream>>>(Wo + (size_t)i * 512 * 512,
                                             woT + (size_t)i * 512 * 512, 512, 512);
    k_transpose_cvt<<<dim3(64, 16), 256, 0, stream>>>(W1 + (size_t)i * 512 * 2048,
                                             w1T + (size_t)i * 2048 * 512, 512, 2048);
    k_transpose_cvt<<<dim3(16, 64), 256, 0, stream>>>(W2 + (size_t)i * 2048 * 512,
                                             w2T + (size_t)i * 512 * 2048, 2048, 512);
  }
  k_transpose_cvt<<<dim3(1000, 16), 256, 0, stream>>>(Wl, wlT, 512, VV);

  // ---- embedding ----
  k_embed<<<A2, 256, 0, stream>>>(x, tok, pos, hf, hb);

  // ---- transformer layers ----
  for (int i = 0; i < DEPTHH; ++i) {
    k_gemm_bf16<<<dim3(1536 / 128, A2 / 128), 256, 0, stream>>>(
        hb, qkvT + (size_t)i * 1536 * 512, qkv, A2, 1536, 512);
    k_pack_qkv<<<dim3(NN / 32, BB * HH), 256, 0, stream>>>(qkv, qb, kbf, vtb);
    k_flash_attn<<<(BB * HH * (NN / 16)) / 8, 256, 0, stream>>>(qb, kbf, vtb, attnb);
    k_gemm_bf16<<<dim3(512 / 128, A2 / 128), 256, 0, stream>>>(
        attnb, woT + (size_t)i * 512 * 512, tmp, A2, 512, 512);
    k_residual_ln<<<A2, 256, 0, stream>>>(tmp, hf, hb,
                                          ln1g + (size_t)i * DD, ln1b + (size_t)i * DD);
    k_gemm_bf16<<<dim3(DFFF / 128, A2 / 128), 256, 0, stream>>>(
        hb, w1T + (size_t)i * 2048 * 512, ff1, A2, DFFF, 512);
    k_gelu_cvt<<<A2 * DFFF / 256, 256, 0, stream>>>(ff1, ffb);
    k_gemm_bf16<<<dim3(512 / 128, A2 / 128), 256, 0, stream>>>(
        ffb, w2T + (size_t)i * 512 * 2048, tmp, A2, 512, DFFF);
    k_residual_ln<<<A2, 256, 0, stream>>>(tmp, hf, hb,
                                          ln2g + (size_t)i * DD, ln2b + (size_t)i * DD);
  }

  // ---- logits ----
  k_gemm_bf16<<<dim3(VV / 128, A2 / 128), 256, 0, stream>>>(hb, wlT, out, A2, VV, 512);
}
